// OneFormerHungarianMatcher_83760452207136
// MI455X (gfx1250) — matrix operand order repaired and checked by reading, not compiled, not run on hardware
//
#include <hip/hip_runtime.h>
#include <hip/hip_bf16.h>
#include <math.h>

#define B_    8
#define Q_    150
#define QPAD_ 160
#define C_    134
#define T_    32
#define H_    160
#define W_    160
#define P_    12544      // divisible by 32 -> all fragment loads 16B aligned
#define KCH_  (P_ / 4)   // 3136, per-wave K chunk (divisible by 32)

typedef __attribute__((ext_vector_type(16))) _Float16 v16h;
typedef __attribute__((ext_vector_type(8)))  _Float16 v8h;
typedef __attribute__((ext_vector_type(8)))  float    v8f;

// ---------------------------------------------------------------- sampling --
__device__ __forceinline__ float fetch_px(const float* __restrict__ img, int ix, int iy) {
  if ((unsigned)ix >= (unsigned)W_ || (unsigned)iy >= (unsigned)H_) return 0.f;
  return img[iy * W_ + ix];
}

__device__ __forceinline__ float bilerp(const float* __restrict__ img, float cx, float cy) {
  // grid_sample, align_corners=False, zero padding
  float x = cx * (float)W_ - 0.5f;
  float y = cy * (float)H_ - 0.5f;
  float x0f = floorf(x), y0f = floorf(y);
  float tx = x - x0f, ty = y - y0f;
  int x0 = (int)x0f, y0 = (int)y0f;
  float v00 = fetch_px(img, x0,     y0);
  float v01 = fetch_px(img, x0 + 1, y0);
  float v10 = fetch_px(img, x0,     y0 + 1);
  float v11 = fetch_px(img, x0 + 1, y0 + 1);
  return v00 * (1.f - tx) * (1.f - ty) + v01 * tx * (1.f - ty)
       + v10 * (1.f - tx) * ty         + v11 * tx * ty;
}

// Phase 1a: tm = sample(mask_labels) -> f16 [B][T][P], plus rowsum(tm) [B][T]
__global__ __launch_bounds__(256) void ofhm_sample_tm(
    const float* __restrict__ mask_labels,  // [B,T,H,W]
    const float* __restrict__ coords,       // [B,P,2]
    _Float16* __restrict__ tm,              // [B,T,P]
    float* __restrict__ tm_sum)             // [B,T]
{
  int bt = blockIdx.x;
  int b  = bt / T_;
  const float* img = mask_labels + (size_t)bt * (H_ * W_);
  const float* crd = coords + (size_t)b * P_ * 2;
  _Float16* dst = tm + (size_t)bt * P_;
  float acc = 0.f;
  for (int p = threadIdx.x; p < P_; p += 256) {
    float v = bilerp(img, crd[2 * p], crd[2 * p + 1]);
    dst[p] = (_Float16)v;
    acc += v;
  }
  __shared__ float red[256];
  red[threadIdx.x] = acc;
  __syncthreads();
  for (int s = 128; s > 0; s >>= 1) {
    if (threadIdx.x < s) red[threadIdx.x] += red[threadIdx.x + s];
    __syncthreads();
  }
  if (threadIdx.x == 0) tm_sum[bt] = red[0];
}

// Phase 1b: pm = sample(pred masks); store -pm and sigmoid(pm) as f16 rows
// (Q padded to 160), plus rowsum(softplus(pm))/P and rowsum(sigmoid(pm)).
__global__ __launch_bounds__(256) void ofhm_sample_pm(
    const float* __restrict__ masks,    // [B,Q,H,W]
    const float* __restrict__ coords,   // [B,P,2]
    _Float16* __restrict__ aneg,        // [B,QPAD,P]  holds -pm
    _Float16* __restrict__ asig,        // [B,QPAD,P]  holds sigmoid(pm)
    float* __restrict__ sp_sum,         // [B,QPAD]    rowsum(softplus(pm))/P
    float* __restrict__ sig_sum)        // [B,QPAD]    rowsum(sigmoid(pm))
{
  int bq = blockIdx.x;
  int b  = bq / QPAD_;
  int q  = bq % QPAD_;
  const float* crd = coords + (size_t)b * P_ * 2;
  _Float16* dn = aneg + (size_t)bq * P_;
  _Float16* ds = asig + (size_t)bq * P_;
  float sp_acc = 0.f, sg_acc = 0.f;
  if (q < Q_) {
    const float* img = masks + ((size_t)b * Q_ + q) * (H_ * W_);
    for (int p = threadIdx.x; p < P_; p += 256) {
      float v   = bilerp(img, crd[2 * p], crd[2 * p + 1]);
      float sig = 1.f / (1.f + __expf(-v));
      dn[p] = (_Float16)(-v);
      ds[p] = (_Float16)sig;
      sp_acc += fmaxf(v, 0.f) + log1pf(__expf(-fabsf(v)));  // stable softplus
      sg_acc += sig;
    }
  } else {
    for (int p = threadIdx.x; p < P_; p += 256) { dn[p] = (_Float16)0.f; ds[p] = (_Float16)0.f; }
  }
  __shared__ float r1[256], r2[256];
  r1[threadIdx.x] = sp_acc;
  r2[threadIdx.x] = sg_acc;
  __syncthreads();
  for (int s = 128; s > 0; s >>= 1) {
    if (threadIdx.x < s) { r1[threadIdx.x] += r1[threadIdx.x + s]; r2[threadIdx.x] += r2[threadIdx.x + s]; }
    __syncthreads();
  }
  if (threadIdx.x == 0) {
    sp_sum[bq]  = r1[0] * (1.f / (float)P_);
    sig_sum[bq] = r2[0];
  }
}

// Phase 1c: cost_class[b][q][t] = -softmax(logits[b,q])[labels[b,t]]
__global__ __launch_bounds__(128) void ofhm_class_cost(
    const float* __restrict__ logits,   // [B,Q,C]
    const int*   __restrict__ labels,   // [B,T]
    float* __restrict__ ccost)          // [B,Q,T]
{
  int q = blockIdx.x, b = blockIdx.y;
  const float* l = logits + ((size_t)b * Q_ + q) * C_;
  __shared__ float red[128];
  float m = -3.0e38f;
  for (int c = threadIdx.x; c < C_; c += 128) m = fmaxf(m, l[c]);
  red[threadIdx.x] = m;
  __syncthreads();
  for (int s = 64; s > 0; s >>= 1) {
    if (threadIdx.x < s) red[threadIdx.x] = fmaxf(red[threadIdx.x], red[threadIdx.x + s]);
    __syncthreads();
  }
  float mx = red[0];
  __syncthreads();
  float sacc = 0.f;
  for (int c = threadIdx.x; c < C_; c += 128) sacc += __expf(l[c] - mx);
  red[threadIdx.x] = sacc;
  __syncthreads();
  for (int s = 64; s > 0; s >>= 1) {
    if (threadIdx.x < s) red[threadIdx.x] += red[threadIdx.x + s];
    __syncthreads();
  }
  float inv = 1.f / red[0];
  if (threadIdx.x < T_) {
    int lab = labels[b * T_ + threadIdx.x];
    ccost[((size_t)b * Q_ + q) * T_ + threadIdx.x] = -__expf(l[lab] - mx) * inv;
  }
}

// Phase 2: WMMA GEMM + fused epilogue.
// Block = 4 waves, each wave handles one K chunk of one 16-row q-tile and both
// 16-col t-tiles (A-fragment reuse). 4 accumulators: {-pm, sig} x {t-tile0, t-tile1}.
// Partials reduced through LDS, wave 0 writes final cost.
__global__ __launch_bounds__(128) void ofhm_wmma_cost(
    const _Float16* __restrict__ aneg,   // [B,QPAD,P]
    const _Float16* __restrict__ asig,   // [B,QPAD,P]
    const _Float16* __restrict__ tm,     // [B,T,P]
    const float* __restrict__ sp_sum,    // [B,QPAD]
    const float* __restrict__ sig_sum,   // [B,QPAD]
    const float* __restrict__ tm_sum,    // [B,T]
    const float* __restrict__ ccost,     // [B,Q,T]
    float* __restrict__ out)             // [B,Q,T]
{
  int qt   = blockIdx.x;            // 0..9  (q tile)
  int b    = blockIdx.y;            // 0..7
  int wave = threadIdx.x >> 5;      // 0..3  (K chunk)
  int lane = threadIdx.x & 31;
  int m    = lane & 15;
  int hi   = lane >> 4;

  // A fragment (16x32 f16): lane = row m, two contiguous 8-half chunks at
  // k = hi*16 and k = hi*16 + 8 within each 32-wide K step.
  const _Float16* An = aneg + ((size_t)b * QPAD_ + qt * 16 + m) * P_ + wave * KCH_ + hi * 16;
  const _Float16* As = asig + ((size_t)b * QPAD_ + qt * 16 + m) * P_ + wave * KCH_ + hi * 16;
  // B fragment (32x16 f16): lane = col n (=m here), lanes 0-15 hold K 0-15,
  // lanes 16-31 hold K 16-31 -> one contiguous 16-half chunk per lane.
  const _Float16* Bt0 = tm + ((size_t)b * T_ + m)      * P_ + wave * KCH_ + hi * 16;
  const _Float16* Bt1 = tm + ((size_t)b * T_ + 16 + m) * P_ + wave * KCH_ + hi * 16;

  v8f c0 = {}, c1 = {}, c2 = {}, c3 = {};
  for (int p0 = 0; p0 < KCH_; p0 += 32) {
    v8h anl = *(const v8h*)(An + p0);
    v8h anh = *(const v8h*)(An + p0 + 8);
    v8h asl = *(const v8h*)(As + p0);
    v8h ash = *(const v8h*)(As + p0 + 8);
    v16h a_n = __builtin_shufflevector(anl, anh, 0,1,2,3,4,5,6,7,8,9,10,11,12,13,14,15);
    v16h a_s = __builtin_shufflevector(asl, ash, 0,1,2,3,4,5,6,7,8,9,10,11,12,13,14,15);
    v16h bf0 = *(const v16h*)(Bt0 + p0);
    v16h bf1 = *(const v16h*)(Bt1 + p0);
    c0 = __builtin_amdgcn_wmma_f32_16x16x32_f16(false, a_n, false, bf0, (short)0, c0, false, false);
    c1 = __builtin_amdgcn_wmma_f32_16x16x32_f16(false, a_n, false, bf1, (short)0, c1, false, false);
    c2 = __builtin_amdgcn_wmma_f32_16x16x32_f16(false, a_s, false, bf0, (short)0, c2, false, false);
    c3 = __builtin_amdgcn_wmma_f32_16x16x32_f16(false, a_s, false, bf1, (short)0, c3, false, false);
  }

  // Reduce the 4 K-chunk partials through LDS.
  __shared__ float part[4][4][8][32];   // [wave][acc][row][lane] = 16 KB
  for (int r = 0; r < 8; ++r) {
    part[wave][0][r][lane] = c0[r];
    part[wave][1][r][lane] = c1[r];
    part[wave][2][r][lane] = c2[r];
    part[wave][3][r][lane] = c3[r];
  }
  __syncthreads();

  if (wave == 0) {
    const float invP = 1.f / (float)P_;
    int n = lane & 15;
    for (int r = 0; r < 8; ++r) {
      // C/D layout: VGPR r -> M = r + hi*8, N = lane&15
      int q = qt * 16 + r + hi * 8;
      if (q >= Q_) continue;
      float sp = sp_sum[b * QPAD_ + q];
      float sg = sig_sum[b * QPAD_ + q];
      float cn0 = part[0][0][r][lane] + part[1][0][r][lane] + part[2][0][r][lane] + part[3][0][r][lane];
      float cn1 = part[0][1][r][lane] + part[1][1][r][lane] + part[2][1][r][lane] + part[3][1][r][lane];
      float cs0 = part[0][2][r][lane] + part[1][2][r][lane] + part[2][2][r][lane] + part[3][2][r][lane];
      float cs1 = part[0][3][r][lane] + part[1][3][r][lane] + part[2][3][r][lane] + part[3][3][r][lane];
      {
        int t = n;
        float cmask = cn0 * invP + sp;                                      // 5.0 weight below
        float dice  = 1.f - (2.f * cs0 + 1.f) / (sg + tm_sum[b * T_ + t] + 1.f);
        float cc    = ccost[((size_t)b * Q_ + q) * T_ + t];
        out[((size_t)b * Q_ + q) * T_ + t] = 5.f * cmask + cc + 5.f * dice;
      }
      {
        int t = 16 + n;
        float cmask = cn1 * invP + sp;
        float dice  = 1.f - (2.f * cs1 + 1.f) / (sg + tm_sum[b * T_ + t] + 1.f);
        float cc    = ccost[((size_t)b * Q_ + q) * T_ + t];
        out[((size_t)b * Q_ + q) * T_ + t] = 5.f * cmask + cc + 5.f * dice;
      }
    }
  }
}

// ---------------------------------------------------------------- launcher --
extern "C" void kernel_launch(void* const* d_in, const int* in_sizes, int n_in,
                              void* d_out, int out_size, void* d_ws, size_t ws_size,
                              hipStream_t stream) {
  const float* masks  = (const float*)d_in[0];  // [B,Q,H,W]
  const float* logits = (const float*)d_in[1];  // [B,Q,C]
  const float* mlab   = (const float*)d_in[2];  // [B,T,H,W]
  const int*   clab   = (const int*)d_in[3];    // [B,T]
  const float* coords = (const float*)d_in[4];  // [B,P,2]
  (void)in_sizes; (void)n_in; (void)out_size; (void)ws_size;

  char* ws = (char*)d_ws;
  size_t off = 0;
  auto take = [&](size_t bytes) -> char* {
    char* p = ws + off;
    off = (off + bytes + 255) & ~(size_t)255;
    return p;
  };
  _Float16* tm      = (_Float16*)take((size_t)B_ * T_ * P_ * sizeof(_Float16));      // 6.4 MB
  _Float16* aneg    = (_Float16*)take((size_t)B_ * QPAD_ * P_ * sizeof(_Float16));   // 32.1 MB
  _Float16* asig    = (_Float16*)take((size_t)B_ * QPAD_ * P_ * sizeof(_Float16));   // 32.1 MB
  float*    tm_sum  = (float*)take((size_t)B_ * T_ * sizeof(float));
  float*    sp_sum  = (float*)take((size_t)B_ * QPAD_ * sizeof(float));
  float*    sig_sum = (float*)take((size_t)B_ * QPAD_ * sizeof(float));
  float*    ccost   = (float*)take((size_t)B_ * Q_ * T_ * sizeof(float));

  ofhm_sample_tm<<<B_ * T_, 256, 0, stream>>>(mlab, coords, tm, tm_sum);
  ofhm_sample_pm<<<B_ * QPAD_, 256, 0, stream>>>(masks, coords, aneg, asig, sp_sum, sig_sum);
  ofhm_class_cost<<<dim3(Q_, B_), 128, 0, stream>>>(logits, clab, ccost);
  ofhm_wmma_cost<<<dim3(QPAD_ / 16, B_), 128, 0, stream>>>(aneg, asig, tm, sp_sum, sig_sum,
                                                           tm_sum, ccost, (float*)d_out);
}
